// DynamicMixtureRNN_40656160424379
// MI455X (gfx1250) — matrix operand order repaired and checked by reading, not compiled, not run on hardware
//
#include <hip/hip_runtime.h>
#include <hip/hip_bf16.h>
#include <math.h>

// ---------------------------------------------------------------------------
// DynamicMixtureRNN on MI455X (gfx1250, wave32).
// Strategy: bf16 V_WMMA_F32_16X16X32_BF16 everywhere (fp32 WMMA is 16x16x4 =
// 8x lower MAC density). One WG per batch element per recurrence step keeps
// the whole 48x512 padded state in ~299KB LDS (< 320KB/WGP).
// ---------------------------------------------------------------------------

#define B_    64
#define T_    512
#define D_    512
#define NV_   32000
#define NROW  33     // 1 z row + 32 mover rows
#define NPAD  48     // padded to 3 WMMA row tiles
#define EPS_  1e-6f

typedef __bf16 bf16;
typedef __attribute__((ext_vector_type(16))) __bf16 v16bf;
typedef __attribute__((ext_vector_type(8)))  __bf16 v8bf;
typedef __attribute__((ext_vector_type(8)))  float  v8f;

// A-fragment (16x32 bf16, row-major source, stride in elements).
// ISA layout: lanes 0-15 hold M=lane with K {0..7,16..23}; lanes 16-31 hold
// M=lane-16 with K {8..15,24..31}  -> two contiguous 16B chunks per lane.
__device__ __forceinline__ v16bf load_a_frag(const bf16* base, int stride,
                                             int m0, int k0) {
  const int lane = threadIdx.x & 31;
  const bf16* p = base + (m0 + (lane & 15)) * stride + k0 + ((lane >> 4) << 3);
  v8bf lo = *(const v8bf*)(p);
  v8bf hi = *(const v8bf*)(p + 16);
  return __builtin_shufflevector(lo, hi, 0, 1, 2, 3, 4, 5, 6, 7,
                                 8, 9, 10, 11, 12, 13, 14, 15);
}

// B-fragment (32x16 bf16) from an N-major ("transposed") array: row n holds
// K contiguously. Lanes 0-15: N=lane, K {0..7,16..23}; lanes 16-31: K {8..15,24..31}.
__device__ __forceinline__ v16bf load_b_fragT(const bf16* baseT, int stride,
                                              int n0, int k0) {
  const int lane = threadIdx.x & 31;
  const bf16* p = baseT + (n0 + (lane & 15)) * stride + k0 + ((lane >> 4) << 3);
  v8bf lo = *(const v8bf*)(p);
  v8bf hi = *(const v8bf*)(p + 16);
  return __builtin_shufflevector(lo, hi, 0, 1, 2, 3, 4, 5, 6, 7,
                                 8, 9, 10, 11, 12, 13, 14, 15);
}

__device__ __forceinline__ v8f wmma_bf16(v16bf a, v16bf b, v8f c) {
  return __builtin_amdgcn_wmma_f32_16x16x32_bf16(false, a, false, b,
                                                 (short)0, c, false, false);
}

// ---------------------------------------------------------------------------
// One-time prep: fp32 (R,C) -> bf16 transposed (C,R)
// ---------------------------------------------------------------------------
__global__ void transpose_to_bf16(const float* __restrict__ src,
                                  bf16* __restrict__ dst, int R, int C) {
  long n = (long)R * C;
  for (long i = (long)blockIdx.x * blockDim.x + threadIdx.x; i < n;
       i += (long)gridDim.x * blockDim.x) {
    long r = i / C, c = i % C;
    dst[c * (long)R + r] = (bf16)src[i];
  }
}

// G[b][r][d], r<NROW: row0 = hidden[b][0]; rows1..16 = hidden[b][1..16];
// rows 17..32 = mover_w[r-17].
__global__ void initg_kernel(const float* __restrict__ hidden,
                             const float* __restrict__ mover_w,
                             float* __restrict__ G) {
  long n = (long)B_ * NROW * D_;
  for (long i = (long)blockIdx.x * blockDim.x + threadIdx.x; i < n;
       i += (long)gridDim.x * blockDim.x) {
    int d = (int)(i % D_);
    long rb = i / D_;
    int r = (int)(rb % NROW);
    int b = (int)(rb / NROW);
    float v = (r <= 16) ? hidden[((long)b * 17 + r) * D_ + d]
                        : mover_w[(long)(r - 17) * D_ + d];
    G[i] = v;
  }
}

// ---------------------------------------------------------------------------
// One recurrence step. grid = B_, block = 256 (8 wave32s), dynamic LDS.
// ---------------------------------------------------------------------------
__global__ __launch_bounds__(256) void step_kernel(
    float* __restrict__ G, const bf16* __restrict__ wqT,
    const bf16* __restrict__ wkT, const bf16* __restrict__ wvT,
    const float* __restrict__ bq, const float* __restrict__ bk,
    const float* __restrict__ bv, const float* __restrict__ gamma,
    const float* __restrict__ beta, const int* __restrict__ seq,
    const float* __restrict__ emb, int t) {
  extern __shared__ char smem[];
  bf16*  gbf = (bf16*)(smem);                  // 48*512 bf16   = 49152 B
  float* gf  = (float*)(smem + 49152);         // 33*512 f32    = 67584 B
  bf16*  qs  = (bf16*)(smem + 116736);         // 48*512 bf16
  bf16*  ks_ = (bf16*)(smem + 165888);         // 48*512 bf16
  bf16*  vT  = (bf16*)(smem + 215040);         // 512*64 bf16   = 65536 B
  float* S   = (float*)(smem + 280576);        // 48*64  f32    = 12288 B
  bf16*  abf = (bf16*)(smem + 292864);         // 48*64  bf16   = 6144 B
                                               // total 299008 B < 320 KB

  const int b = blockIdx.x;
  const int tid = threadIdx.x;
  const int wid = tid >> 5;
  const int lane = tid & 31;

  const int tok = seq[(long)b * T_ + t];
  const float* xt = emb + (long)tok * D_;
  float* Gb = G + (long)b * NROW * D_;

  // Phase 1: load carry, add x_t into z row, make fp32 + bf16 copies.
  for (int i = tid; i < NROW * D_; i += 256) {
    int r = i >> 9, d = i & 511;
    float v = Gb[i] + (r == 0 ? xt[d] : 0.f);
    gf[i]  = v;
    gbf[i] = (bf16)v;
  }
  for (int i = tid; i < (NPAD - NROW) * D_; i += 256)
    gbf[NROW * D_ + i] = (bf16)0.f;
  __syncthreads();

  // Phase 2: Q,K,V = G @ W + b.  3 proj x 3 mtiles x 32 ntiles = 288 tiles.
  const float qscale = 0.04419417382415922f;  // 1/sqrt(D)
  for (int job = wid; job < 288; job += 8) {
    int proj = job / 96;
    int rem = job % 96;
    int mt = rem >> 5, nt = rem & 31;
    const bf16* WT = (proj == 0) ? wqT : (proj == 1) ? wkT : wvT;
    const float* bias = (proj == 0) ? bq : (proj == 1) ? bk : bv;
    float bb = bias[nt * 16 + (lane & 15)];
    v8f acc = {bb, bb, bb, bb, bb, bb, bb, bb};
#pragma unroll 4
    for (int k0 = 0; k0 < D_; k0 += 32) {
      v16bf a = load_a_frag(gbf, D_, mt * 16, k0);
      v16bf w = load_b_fragT(WT, D_, nt * 16, k0);
      acc = wmma_bf16(a, w, acc);
    }
    int n = nt * 16 + (lane & 15);
    int mb = mt * 16 + ((lane >> 4) << 3);
    if (proj == 2) {
      // store V transposed so B-fragments of A@V are contiguous
      v8bf pk;
#pragma unroll
      for (int r = 0; r < 8; ++r) pk[r] = (bf16)acc[r];
      *(v8bf*)(vT + n * 64 + mb) = pk;
    } else {
      bf16* dst = (proj == 0) ? qs : ks_;
      float sc = (proj == 0) ? qscale : 1.f;
#pragma unroll
      for (int r = 0; r < 8; ++r) dst[(mb + r) * D_ + n] = (bf16)(acc[r] * sc);
    }
  }
  __syncthreads();

  // Phase 3: S = (Q/sqrt(D)) @ K^T.  K row-major == K^T in B layout.
  for (int job = wid; job < 9; job += 8) {
    int mt = job / 3, nt = job % 3;
    v8f acc = {};
#pragma unroll 4
    for (int k0 = 0; k0 < D_; k0 += 32) {
      v16bf a = load_a_frag(qs, D_, mt * 16, k0);
      v16bf kk = load_b_fragT(ks_, D_, nt * 16, k0);
      acc = wmma_bf16(a, kk, acc);
    }
    int n = nt * 16 + (lane & 15);
    int mb = mt * 16 + ((lane >> 4) << 3);
#pragma unroll
    for (int r = 0; r < 8; ++r) S[(mb + r) * 64 + n] = acc[r];
  }
  __syncthreads();

  // Phase 4: softmax over the 33 valid keys; pad cols/rows -> 0 probs.
  if (tid < NPAD) {
    int r = tid;
    if (r < NROW) {
      float mx = -1e30f;
      for (int c = 0; c < NROW; ++c) mx = fmaxf(mx, S[r * 64 + c]);
      float sum = 0.f;
      for (int c = 0; c < NROW; ++c) sum += __expf(S[r * 64 + c] - mx);
      float inv = 1.f / sum;
      for (int c = 0; c < 64; ++c)
        abf[r * 64 + c] =
            (c < NROW) ? (bf16)(__expf(S[r * 64 + c] - mx) * inv) : (bf16)0.f;
    } else {
      for (int c = 0; c < 64; ++c) abf[r * 64 + c] = (bf16)0.f;
    }
  }
  __syncthreads();

  // Phase 5: dg = A @ V (K=64 padded), residual h = g + 2*dg into gf.
  for (int job = wid; job < 96; job += 8) {
    int mt = job >> 5, nt = job & 31;
    v8f acc = {};
#pragma unroll
    for (int k0 = 0; k0 < 64; k0 += 32) {
      v16bf a = load_a_frag(abf, 64, mt * 16, k0);
      v16bf v = load_b_fragT(vT, 64, nt * 16, k0);
      acc = wmma_bf16(a, v, acc);
    }
    int n = nt * 16 + (lane & 15);
    int mb = mt * 16 + ((lane >> 4) << 3);
#pragma unroll
    for (int r = 0; r < 8; ++r) {
      int m = mb + r;
      if (m < NROW) gf[m * D_ + n] += 2.f * acc[r];
    }
  }
  __syncthreads();

  // Phase 6: layernorm each valid row (wave32 shuffle reductions), store carry.
  for (int r = wid; r < NROW; r += 8) {
    float s = 0.f;
    for (int d = lane; d < D_; d += 32) s += gf[r * D_ + d];
#pragma unroll
    for (int off = 16; off > 0; off >>= 1) s += __shfl_xor(s, off, 32);
    float mu = s * (1.f / D_);
    float vs = 0.f;
    for (int d = lane; d < D_; d += 32) {
      float x = gf[r * D_ + d] - mu;
      vs += x * x;
    }
#pragma unroll
    for (int off = 16; off > 0; off >>= 1) vs += __shfl_xor(vs, off, 32);
    float inv = rsqrtf(vs * (1.f / D_) + EPS_);
    for (int d = lane; d < D_; d += 32)
      Gb[r * D_ + d] = (gf[r * D_ + d] - mu) * inv * gamma[d] + beta[d];
  }
}

// ---------------------------------------------------------------------------
// Finish: z -> bf16, logits = z @ vocab_W + b (WMMA), log-softmax, copy g out.
// ---------------------------------------------------------------------------
__global__ void zcast_kernel(const float* __restrict__ G,
                             bf16* __restrict__ zbf) {
  int i = blockIdx.x * blockDim.x + threadIdx.x;
  if (i < B_ * D_) {
    int b = i >> 9, d = i & 511;
    zbf[i] = (bf16)G[((long)b * NROW) * D_ + d];
  }
}

__global__ __launch_bounds__(256) void logits_kernel(
    const bf16* __restrict__ zbf, const bf16* __restrict__ vocT,
    const float* __restrict__ vb, float* __restrict__ logits) {
  const int blk = blockIdx.x;  // NV_/128 blocks, each 64x128 output tile
  const int wid = threadIdx.x >> 5;
  const int lane = threadIdx.x & 31;
  for (int job = wid; job < 32; job += 8) {
    int mt = job >> 3, nt = job & 7;
    int n0 = blk * 128 + nt * 16;
    float bb = vb[n0 + (lane & 15)];
    v8f acc = {bb, bb, bb, bb, bb, bb, bb, bb};
#pragma unroll 4
    for (int k0 = 0; k0 < D_; k0 += 32) {
      v16bf a = load_a_frag(zbf, D_, mt * 16, k0);
      v16bf w = load_b_fragT(vocT, D_, n0, k0);
      acc = wmma_bf16(a, w, acc);
    }
    int n = n0 + (lane & 15);
    int mb = mt * 16 + ((lane >> 4) << 3);
#pragma unroll
    for (int r = 0; r < 8; ++r) logits[(long)(mb + r) * NV_ + n] = acc[r];
  }
}

__global__ __launch_bounds__(256) void lsm_kernel(const float* __restrict__ L0,
                                                  float* __restrict__ y) {
  __shared__ float red[256];
  const int b = blockIdx.x, tid = threadIdx.x;
  const float* L = L0 + (long)b * NV_;
  float mx = -1e30f;
  for (int i = tid; i < NV_; i += 256) mx = fmaxf(mx, L[i]);
  red[tid] = mx;
  __syncthreads();
  for (int s = 128; s > 0; s >>= 1) {
    if (tid < s) red[tid] = fmaxf(red[tid], red[tid + s]);
    __syncthreads();
  }
  mx = red[0];
  __syncthreads();
  float sum = 0.f;
  for (int i = tid; i < NV_; i += 256) sum += __expf(L[i] - mx);
  red[tid] = sum;
  __syncthreads();
  for (int s = 128; s > 0; s >>= 1) {
    if (tid < s) red[tid] += red[tid + s];
    __syncthreads();
  }
  float lse = mx + logf(red[0]);
  for (int i = tid; i < NV_; i += 256) y[(long)b * NV_ + i] = L[i] - lse;
}

__global__ void outg_kernel(const float* __restrict__ G,
                            float* __restrict__ out) {
  int i = blockIdx.x * blockDim.x + threadIdx.x;
  if (i < B_ * 17 * D_) {
    int b = i / (17 * D_);
    int rd = i % (17 * D_);  // rows 0..16 are contiguous within a batch
    out[i] = G[((long)b * NROW) * D_ + rd];
  }
}

// ---------------------------------------------------------------------------
extern "C" void kernel_launch(void* const* d_in, const int* in_sizes, int n_in,
                              void* d_out, int out_size, void* d_ws,
                              size_t ws_size, hipStream_t stream) {
  const float* hidden  = (const float*)d_in[0];
  const int*   seq     = (const int*)d_in[1];
  const float* emb     = (const float*)d_in[2];
  const float* Wq      = (const float*)d_in[3];
  const float* bq      = (const float*)d_in[4];
  const float* Wk      = (const float*)d_in[5];
  const float* bk      = (const float*)d_in[6];
  const float* Wv      = (const float*)d_in[7];
  const float* bv      = (const float*)d_in[8];
  const float* mover_w = (const float*)d_in[9];
  const float* gamma   = (const float*)d_in[10];
  const float* beta    = (const float*)d_in[11];
  const float* vocab_W = (const float*)d_in[12];
  const float* vocab_b = (const float*)d_in[13];
  float* out = (float*)d_out;

  char* ws = (char*)d_ws;
  size_t off = 0;
  auto carve = [&](size_t bytes) -> void* {
    void* p = ws + off;
    off = (off + bytes + 255) & ~(size_t)255;
    return p;
  };
  bf16*  wqT    = (bf16*)carve((size_t)D_ * D_ * sizeof(bf16));
  bf16*  wkT    = (bf16*)carve((size_t)D_ * D_ * sizeof(bf16));
  bf16*  wvT    = (bf16*)carve((size_t)D_ * D_ * sizeof(bf16));
  bf16*  vocT   = (bf16*)carve((size_t)NV_ * D_ * sizeof(bf16));
  float* G      = (float*)carve((size_t)B_ * NROW * D_ * sizeof(float));
  bf16*  zbf    = (bf16*)carve((size_t)B_ * D_ * sizeof(bf16));
  float* logits = (float*)carve((size_t)B_ * NV_ * sizeof(float));

  // Prep: bf16 transposed weights (B-fragment friendly, N-major).
  transpose_to_bf16<<<512, 256, 0, stream>>>(Wq, wqT, D_, D_);
  transpose_to_bf16<<<512, 256, 0, stream>>>(Wk, wkT, D_, D_);
  transpose_to_bf16<<<512, 256, 0, stream>>>(Wv, wvT, D_, D_);
  transpose_to_bf16<<<4096, 256, 0, stream>>>(vocab_W, vocT, D_, NV_);
  initg_kernel<<<2048, 256, 0, stream>>>(hidden, mover_w, G);

  const size_t STEP_LDS = 299008;  // see layout in step_kernel
  hipFuncSetAttribute(reinterpret_cast<const void*>(step_kernel),
                      hipFuncAttributeMaxDynamicSharedMemorySize,
                      (int)STEP_LDS);
  for (int t = 0; t < T_; ++t)
    step_kernel<<<B_, 256, STEP_LDS, stream>>>(G, wqT, wkT, wvT, bq, bk, bv,
                                               gamma, beta, seq, emb, t);

  zcast_kernel<<<(B_ * D_ + 255) / 256, 256, 0, stream>>>(G, zbf);
  logits_kernel<<<NV_ / 128, 256, 0, stream>>>(zbf, vocT, vocab_b, logits);
  lsm_kernel<<<B_, 256, 0, stream>>>(logits, out + (size_t)B_ * 17 * D_);
  outg_kernel<<<(B_ * 17 * D_ + 255) / 256, 256, 0, stream>>>(G, out);
}
